// Model_32169305047208
// MI455X (gfx1250) — matrix operand order repaired and checked by reading, not compiled, not run on hardware
//
#include <hip/hip_runtime.h>
#include <stdint.h>

// ---------------------------------------------------------------------------
// MI455X (gfx1250) implementation of the FEDformer-style reference.
// All matmuls (incl. rFFT/irFFT recast as dense DFT GEMMs, since only 64
// modes survive) run on v_wmma_f32_16x16x32_bf16 (bf16 in, f32 accumulate).
// Operand layouts are swizzled so every WMMA fragment is a single contiguous
// 32-byte load; GEMMs are double-buffered through LDS (1 barrier / K-tile).
// ---------------------------------------------------------------------------

typedef __attribute__((ext_vector_type(16))) __bf16 v16bf;
typedef __attribute__((ext_vector_type(8)))  __bf16 v8bf;
typedef __attribute__((ext_vector_type(2)))  __bf16 v2bf;
typedef __attribute__((ext_vector_type(8)))  float  v8f;

#define B_    32
#define L_    1024
#define CIN_  21
#define D_    512
#define H_    8
#define E_    64
#define DFF_  2048
#define MOD_  64
#define MROWS 32768   /* B*L  */
#define SIG   16384   /* B*H*E */

// A-fragment K-swizzle: 16-bit WMMA A wants lanes<16 to hold K{0-7,16-23} and
// lanes>=16 to hold K{8-15,24-31}. Storing K with bits 3<->4 swapped makes each
// lane's fragment contiguous.
static __device__ __forceinline__ int swz32(int k) {
  return (k & ~0x18) | ((k & 8) << 1) | ((k & 16) >> 1);
}

// ---------------- epilogue modes for the GEMMs -----------------------------
constexpr int EP_F32     = 0;  // Cf = acc (+bias)
constexpr int EP_RESID   = 1;  // Cf = acc (+bias) + resid
constexpr int EP_GELU_BF = 2;  // Cbf = bf16(gelu(acc))
constexpr int EP_BF      = 3;  // Cbf = bf16(acc)
constexpr int EP_QSCAT   = 4;  // scatter to [B, H*E, L] bf16 (Fourier layout)
constexpr int EP_XSCAT   = 5;  // scatter spectrum to [h][m][re|im][b][e]

template<int MODE>
static __device__ __forceinline__ void ep_store(float v, int row, int col, int N,
    const float* __restrict__ bias, const float* __restrict__ resid,
    float* __restrict__ Cf, __bf16* __restrict__ Cbf)
{
  if (bias) v += bias[col];
  if (MODE == EP_F32) {
    Cf[(size_t)row * N + col] = v;
  } else if (MODE == EP_RESID) {
    Cf[(size_t)row * N + col] = v + resid[(size_t)row * N + col];
  } else if (MODE == EP_GELU_BF) {
    v = 0.5f * v * (1.0f + erff(v * 0.70710678118f));
    Cbf[(size_t)row * N + col] = (__bf16)v;
  } else if (MODE == EP_BF) {
    Cbf[(size_t)row * N + col] = (__bf16)v;
  } else if (MODE == EP_QSCAT) {     // row = b*1024 + l ; col = h*64+e
    int b = row >> 10, l = row & 1023;
    Cbf[((size_t)b * D_ + col) * L_ + l] = (__bf16)v;
  } else {                           // EP_XSCAT: row = b*512+h*64+e ; col = 2m+pl
    int b = row >> 9, h = (row >> 6) & 7, e = row & 63;
    int mm = col >> 1, pl = col & 1;
    int es = e;
    Cbf[(((size_t)(h * MOD_ + mm) * 2 + pl) * B_ + b) * E_ + es] = (__bf16)v;
  }
}

// ---------------------------------------------------------------------------
// Big GEMM: C[M,N] = A[M,K] * W[N,K]^T ; block tile 128(M) x 256(N), Kstep 32.
// 8 waves in 2(M) x 4(N); each wave 64x64 = 4x4 WMMA accums (16 WMMA per
// K-tile against 8 fragment loads). Double-buffered LDS, 1 barrier / K-tile.
// ---------------------------------------------------------------------------
template<int MODE>
__global__ __launch_bounds__(256)
void gemm_big(const __bf16* __restrict__ A, const __bf16* __restrict__ W,
              const float* __restrict__ bias, const float* __restrict__ resid,
              float* __restrict__ Cf, __bf16* __restrict__ Cbf,
              int M, int N, int K)
{
  __shared__ __align__(64) __bf16 smA[2][128 * 32];
  __shared__ __align__(64) __bf16 smB[2][256 * 32];
  const int bm   = blockIdx.y * 128;
  const int bn   = blockIdx.x * 256;
  const int tid  = threadIdx.x;
  const int lane = tid & 31, wave = tid >> 5;
  const int l16  = lane & 15, lh = lane >> 4;
  const int waveM = (wave >> 2) * 64;
  const int waveN = (wave & 3) * 64;
  const int ldrA = tid >> 1;          // A loader row
  const int ldcA = (tid & 1) * 16;    // A loader K half
  const int posA = ldcA >> 1;         // swizzled LDS slot (0 or 8)

  const __bf16* gA = A + (size_t)(bm + ldrA) * K + ldcA;
  const __bf16* gB = W + (size_t)(bn + tid) * K;

  v8f acc[4][4] = {};

  // preload K-tile 0
  {
    v8bf a0 = *(const v8bf*)gA, a1 = *(const v8bf*)(gA + 8);
    v8bf b0 = *(const v8bf*)gB,        b1 = *(const v8bf*)(gB + 8);
    v8bf b2 = *(const v8bf*)(gB + 16), b3 = *(const v8bf*)(gB + 24);
    *(v8bf*)&smA[0][ldrA * 32 + posA]      = a0;   // swz: k16-23 -> pos 8
    *(v8bf*)&smA[0][ldrA * 32 + posA + 16] = a1;
    *(v8bf*)&smB[0][tid * 32 + 0]  = b0;
    *(v8bf*)&smB[0][tid * 32 + 16] = b1;
    *(v8bf*)&smB[0][tid * 32 + 8]  = b2;
    *(v8bf*)&smB[0][tid * 32 + 24] = b3;
  }
  __syncthreads();

  const int nk = K >> 5;
  for (int t = 0; t < nk; ++t) {
    const int cur = t & 1, nxt = cur ^ 1;
    const bool has = (t + 1 < nk);
    v8bf a0, a1, b0, b1, b2, b3;
    if (has) {                        // stage next tile while WMMA runs
      const __bf16* pa = gA + (size_t)(t + 1) * 32;
      a0 = *(const v8bf*)pa; a1 = *(const v8bf*)(pa + 8);
      const __bf16* pb = gB + (size_t)(t + 1) * 32;
      b0 = *(const v8bf*)pb;        b1 = *(const v8bf*)(pb + 8);
      b2 = *(const v8bf*)(pb + 16); b3 = *(const v8bf*)(pb + 24);
      if (t + 2 < nk) {               // gfx1250 global_prefetch for tile t+2
        __builtin_prefetch(gA + (size_t)(t + 2) * 32, 0, 3);
        __builtin_prefetch(gB + (size_t)(t + 2) * 32, 0, 3);
      }
    }

    v16bf af[4];
#pragma unroll
    for (int i = 0; i < 4; ++i)
      af[i] = *(const v16bf*)&smA[cur][(waveM + i * 16 + l16) * 32 + lh * 16];
    v16bf bf4[4];
#pragma unroll
    for (int j = 0; j < 4; ++j)
      bf4[j] = *(const v16bf*)&smB[cur][(waveN + j * 16 + l16) * 32 + lh * 16];

#pragma unroll
    for (int i = 0; i < 4; ++i)
#pragma unroll
      for (int j = 0; j < 4; ++j)
        acc[i][j] = __builtin_amdgcn_wmma_f32_16x16x32_bf16(
            false, af[i], false, bf4[j], (short)0, acc[i][j], false, false);

    if (has) {
      *(v8bf*)&smA[nxt][ldrA * 32 + posA]      = a0;
      *(v8bf*)&smA[nxt][ldrA * 32 + posA + 16] = a1;
      *(v8bf*)&smB[nxt][tid * 32 + 0]  = b0;
      *(v8bf*)&smB[nxt][tid * 32 + 16] = b1;
      *(v8bf*)&smB[nxt][tid * 32 + 8]  = b2;
      *(v8bf*)&smB[nxt][tid * 32 + 24] = b3;
    }
    __syncthreads();
  }

#pragma unroll
  for (int i = 0; i < 4; ++i)
#pragma unroll
    for (int j = 0; j < 4; ++j)
#pragma unroll
      for (int r = 0; r < 8; ++r) {
        int row = bm + waveM + i * 16 + lh * 8 + r;   // C/D: M = r + 8*(lane>>4)
        int col = bn + waveN + j * 16 + l16;
        ep_store<MODE>(acc[i][j][r], row, col, N, bias, resid, Cf, Cbf);
      }
}

// ---------------------------------------------------------------------------
// Narrow GEMM (N=128 rFFT projection): 128x128 block, wave 64x32 (4x2),
// double-buffered LDS.
// ---------------------------------------------------------------------------
template<int MODE>
__global__ __launch_bounds__(256)
void gemm_k128(const __bf16* __restrict__ A, const __bf16* __restrict__ W,
               const float* __restrict__ bias, const float* __restrict__ resid,
               float* __restrict__ Cf, __bf16* __restrict__ Cbf,
               int M, int N, int K)
{
  __shared__ __align__(64) __bf16 smA[2][128 * 32];
  __shared__ __align__(64) __bf16 smB[2][128 * 32];
  const int bm   = blockIdx.y * 128;
  const int bn   = blockIdx.x * 128;
  const int tid  = threadIdx.x;
  const int lane = tid & 31, wave = tid >> 5;
  const int l16  = lane & 15, lh = lane >> 4;
  const int waveM = (wave >> 2) * 64;
  const int waveN = (wave & 3) * 32;
  const int ldr = tid >> 1;
  const int ldc = (tid & 1) * 16;
  const int posA = ldc >> 1;

  const __bf16* gA = A + (size_t)(bm + ldr) * K + ldc;
  const __bf16* gW = W + (size_t)(bn + ldr) * K + ldc;

  v8f acc[4][2] = {};

  {
    v8bf a0 = *(const v8bf*)gA, a1 = *(const v8bf*)(gA + 8);
    v8bf w0 = *(const v8bf*)gW, w1 = *(const v8bf*)(gW + 8);
    *(v8bf*)&smA[0][ldr * 32 + posA]      = a0;
    *(v8bf*)&smA[0][ldr * 32 + posA + 16] = a1;
    *(v8bf*)&smB[0][ldr * 32 + posA]      = w0;
    *(v8bf*)&smB[0][ldr * 32 + posA + 16] = w1;
  }
  __syncthreads();

  const int nk = K >> 5;
  for (int t = 0; t < nk; ++t) {
    const int cur = t & 1, nxt = cur ^ 1;
    const bool has = (t + 1 < nk);
    v8bf a0, a1, w0, w1;
    if (has) {
      const __bf16* pa = gA + (size_t)(t + 1) * 32;
      a0 = *(const v8bf*)pa; a1 = *(const v8bf*)(pa + 8);
      const __bf16* pw = gW + (size_t)(t + 1) * 32;
      w0 = *(const v8bf*)pw; w1 = *(const v8bf*)(pw + 8);
      if (t + 2 < nk) {
        __builtin_prefetch(gA + (size_t)(t + 2) * 32, 0, 3);
        __builtin_prefetch(gW + (size_t)(t + 2) * 32, 0, 3);
      }
    }

    v16bf af[4];
#pragma unroll
    for (int i = 0; i < 4; ++i)
      af[i] = *(const v16bf*)&smA[cur][(waveM + i * 16 + l16) * 32 + lh * 16];
    v16bf bf2[2];
#pragma unroll
    for (int j = 0; j < 2; ++j)
      bf2[j] = *(const v16bf*)&smB[cur][(waveN + j * 16 + l16) * 32 + lh * 16];

#pragma unroll
    for (int i = 0; i < 4; ++i)
#pragma unroll
      for (int j = 0; j < 2; ++j)
        acc[i][j] = __builtin_amdgcn_wmma_f32_16x16x32_bf16(
            false, af[i], false, bf2[j], (short)0, acc[i][j], false, false);

    if (has) {
      *(v8bf*)&smA[nxt][ldr * 32 + posA]      = a0;
      *(v8bf*)&smA[nxt][ldr * 32 + posA + 16] = a1;
      *(v8bf*)&smB[nxt][ldr * 32 + posA]      = w0;
      *(v8bf*)&smB[nxt][ldr * 32 + posA + 16] = w1;
    }
    __syncthreads();
  }

#pragma unroll
  for (int i = 0; i < 4; ++i)
#pragma unroll
    for (int j = 0; j < 2; ++j)
#pragma unroll
      for (int r = 0; r < 8; ++r) {
        int row = bm + waveM + i * 16 + lh * 8 + r;
        int col = bn + waveN + j * 16 + l16;
        ep_store<MODE>(acc[i][j][r], row, col, N, bias, resid, Cf, Cbf);
      }
}

// ---------------- per-(h,mode) complex channel mixing ----------------------
// out[b,o] = sum_e X[b,e] * w[e,o]  (complex), B=32, E=64, one wave per block.
// Re += Are*Wre + Aim*(-Wim) ; Im += Are*Wim + Aim*Wre (neg baked into Wimn).
// XF2 layout [h][m][re|im][b][e] and fwT [h][m][o][e] make every
// fragment a single contiguous 32-byte global load.
__global__ __launch_bounds__(32)
void fourier_mix(const __bf16* __restrict__ XF2, const __bf16* __restrict__ Wre,
                 const __bf16* __restrict__ Wim, const __bf16* __restrict__ Wimn,
                 __bf16* __restrict__ YF)
{
  const int m = blockIdx.x;            // mode 0..63
  const int h = blockIdx.y;            // head 0..7
  const int lane = threadIdx.x;
  const int l16 = lane & 15, lh = lane >> 4;
  const size_t baseRe = (size_t)(h * MOD_ + m) * 2 * (B_ * E_);
  const size_t baseIm = baseRe + (B_ * E_);
  const size_t wbase  = (size_t)(h * MOD_ + m) * (E_ * E_);
  v8f accR[2][4] = {};
  v8f accI[2][4] = {};

#pragma unroll
  for (int s = 0; s < 2; ++s) {        // K = 64 in two k=32 steps
    v16bf are[2], aim[2];
#pragma unroll
    for (int i = 0; i < 2; ++i) {
      int off = (i * 16 + l16) * E_ + s * 32 + lh * 16;
      are[i] = *(const v16bf*)&XF2[baseRe + off];
      aim[i] = *(const v16bf*)&XF2[baseIm + off];
    }
#pragma unroll
    for (int j = 0; j < 4; ++j) {
      int woff = (j * 16 + l16) * E_ + s * 32 + lh * 16;
      v16bf wre = *(const v16bf*)&Wre[wbase + woff];
      v16bf wim = *(const v16bf*)&Wim[wbase + woff];
      v16bf wmn = *(const v16bf*)&Wimn[wbase + woff];
#pragma unroll
      for (int i = 0; i < 2; ++i) {
        accR[i][j] = __builtin_amdgcn_wmma_f32_16x16x32_bf16(false, are[i], false, wre, (short)0, accR[i][j], false, false);
        accR[i][j] = __builtin_amdgcn_wmma_f32_16x16x32_bf16(false, aim[i], false, wmn, (short)0, accR[i][j], false, false);
        accI[i][j] = __builtin_amdgcn_wmma_f32_16x16x32_bf16(false, are[i], false, wim, (short)0, accI[i][j], false, false);
        accI[i][j] = __builtin_amdgcn_wmma_f32_16x16x32_bf16(false, aim[i], false, wre, (short)0, accI[i][j], false, false);
      }
    }
  }
  // YF natural [row=(b*512+h*64+o), 128] re/im interleaved (irFFT GEMM A input)
#pragma unroll
  for (int i = 0; i < 2; ++i)
#pragma unroll
    for (int j = 0; j < 4; ++j)
#pragma unroll
      for (int r = 0; r < 8; ++r) {
        int brow = i * 16 + lh * 8 + r;
        int o = j * 16 + l16;
        size_t idx = ((size_t)(brow * H_ + h) * E_ + o) * 128 + 2 * m;
        v2bf pr; pr[0] = (__bf16)accR[i][j][r]; pr[1] = (__bf16)accI[i][j][r];
        *(v2bf*)&YF[idx] = pr;
      }
}

// ---------------- Fourier weight transpose: [h][e][o][m] -> [h][m][o][e] ---
__global__ void fw_transpose(const float* __restrict__ fre, const float* __restrict__ fim,
                             __bf16* __restrict__ tre, __bf16* __restrict__ tim,
                             __bf16* __restrict__ timn)
{
  const int n = 2 * H_ * MOD_ * E_ * E_;
  for (int i = blockIdx.x * blockDim.x + threadIdx.x; i < n; i += gridDim.x * blockDim.x) {
    int e  = i & 63;
    int o  = (i >> 6) & 63;
    int mm = (i >> 12) & 63;
    int lh2 = i >> 18;   // layer*8 + h
    size_t src = (((size_t)lh2 * E_ + e) * E_ + o) * MOD_ + mm;
    float re = fre[src], im = fim[src];
    tre[i] = (__bf16)re; tim[i] = (__bf16)im; timn[i] = (__bf16)(-im);
  }
}

// ---------------- embedding: circular conv1d(k=3) + sinusoidal PE ----------
__global__ __launch_bounds__(256)
void embed(const float* __restrict__ x_enc, const float* __restrict__ conv_w,
           float* __restrict__ X, __bf16* __restrict__ Xbf)
{
  __shared__ float sv[3 * CIN_];
  const int l = blockIdx.x, b = blockIdx.y;
  const int t = threadIdx.x;
  if (t < 3 * CIN_) {
    int j = t / CIN_, c = t % CIN_;
    int row = (l + j + L_ - 1) & (L_ - 1);               // wrap pad
    sv[t] = x_enc[((size_t)b * L_ + row) * CIN_ + c];
  }
  __syncthreads();
  for (int d = t; d < D_; d += 256) {
    float acc = 0.f;
#pragma unroll
    for (int j = 0; j < 3; ++j)
      for (int c = 0; c < CIN_; ++c)
        acc += conv_w[(d * CIN_ + c) * 3 + j] * sv[j * CIN_ + c];
    int i2 = d & ~1;
    float f = __expf((float)i2 * (-9.210340371976184f / (float)D_));
    float sn, cs; __sincosf((float)l * f, &sn, &cs);
    acc += (d & 1) ? cs : sn;
    size_t o = ((size_t)b * L_ + l) * D_ + d;
    X[o] = acc; Xbf[o] = (__bf16)acc;
  }
}

// ---------------- series decomp: x - moving_avg(x, 25), replicate pad ------
__global__ __launch_bounds__(256)
void movavg_sub(const float* __restrict__ X, float* __restrict__ Y,
                __bf16* __restrict__ Ybf)
{
  __shared__ float s[L_];
  const int b = blockIdx.x >> 9;
  const int d = blockIdx.x & (D_ - 1);
  const size_t base = (size_t)b * L_ * D_ + d;
  for (int l = threadIdx.x; l < L_; l += 256) s[l] = X[base + (size_t)l * D_];
  __syncthreads();
  for (int l = threadIdx.x; l < L_; l += 256) {
    float sum = 0.f;
#pragma unroll
    for (int j = -12; j <= 12; ++j) {
      int idx = l + j; idx = idx < 0 ? 0 : (idx > L_ - 1 ? L_ - 1 : idx);
      sum += s[idx];
    }
    float v = s[l] - sum * (1.0f / 25.0f);
    Y[base + (size_t)l * D_] = v;
    Ybf[base + (size_t)l * D_] = (__bf16)v;
  }
}

// ---------------- row LayerNorm over D=512 ---------------------------------
__global__ __launch_bounds__(256)
void ln_rows(const float* __restrict__ X, const float* __restrict__ g,
             const float* __restrict__ bta, float* __restrict__ Y)
{
  __shared__ float red[256];
  const size_t row = blockIdx.x;
  const float* x = X + row * D_;
  float s = 0.f;
  for (int i = threadIdx.x; i < D_; i += 256) s += x[i];
  red[threadIdx.x] = s; __syncthreads();
  for (int o = 128; o > 0; o >>= 1) { if ((int)threadIdx.x < o) red[threadIdx.x] += red[threadIdx.x + o]; __syncthreads(); }
  float mu = red[0] * (1.0f / D_);
  __syncthreads();
  float vs = 0.f;
  for (int i = threadIdx.x; i < D_; i += 256) { float dd = x[i] - mu; vs += dd * dd; }
  red[threadIdx.x] = vs; __syncthreads();
  for (int o = 128; o > 0; o >>= 1) { if ((int)threadIdx.x < o) red[threadIdx.x] += red[threadIdx.x + o]; __syncthreads(); }
  float rs = rsqrtf(red[0] * (1.0f / D_) + 1e-5f);
  for (int i = threadIdx.x; i < D_; i += 256)
    Y[row * D_ + i] = (x[i] - mu) * rs * g[i] + bta[i];
}

// ---------------- subtract per-(b,d) mean over L; emit f32 + bf16 ----------
__global__ __launch_bounds__(256)
void colmean_sub(const float* __restrict__ X, float* __restrict__ Y,
                 __bf16* __restrict__ Ybf)
{
  const int b = blockIdx.y;
  const int d = blockIdx.x * 256 + threadIdx.x;
  const size_t base = (size_t)b * L_ * D_ + d;
  float acc = 0.f;
  for (int l = 0; l < L_; ++l) acc += X[base + (size_t)l * D_];
  float mu = acc * (1.0f / L_);
  for (int l = 0; l < L_; ++l) {
    float v = X[base + (size_t)l * D_] - mu;
    Y[base + (size_t)l * D_] = v;
    Ybf[base + (size_t)l * D_] = (__bf16)v;
  }
}

// ---------------- final regression: out[b] = dot(z[b,:], w) + bias ---------
__global__ __launch_bounds__(256)
void regress(const float* __restrict__ Z, const float* __restrict__ w,
             const float* __restrict__ wb, float* __restrict__ out)
{
  __shared__ float red[256];
  const int b = blockIdx.x;
  const float* z = Z + (size_t)b * (L_ * D_);
  float acc = 0.f;
  for (int i = threadIdx.x; i < L_ * D_; i += 256) acc += z[i] * w[i];
  red[threadIdx.x] = acc; __syncthreads();
  for (int o = 128; o > 0; o >>= 1) { if ((int)threadIdx.x < o) red[threadIdx.x] += red[threadIdx.x + o]; __syncthreads(); }
  if (threadIdx.x == 0) out[b] = red[0] + wb[0];
}

// ---------------- weight f32 -> bf16 ---------------------------------------
__global__ void f32_to_bf16(const float* __restrict__ src, __bf16* __restrict__ dst, int n)
{
  for (int i = blockIdx.x * blockDim.x + threadIdx.x; i < n; i += gridDim.x * blockDim.x)
    dst[i] = (__bf16)src[i];
}

// ---------------- DFT / IDFT bases (exact periodic phase reduction) --------
__global__ void build_basis(__bf16* __restrict__ dft, __bf16* __restrict__ idft)
{
  const float W = 6.283185307179586f / (float)L_;
  int idx = blockIdx.x * blockDim.x + threadIdx.x;
  if (idx >= 128 * L_) return;
  {
    int n = idx >> 10, k = idx & (L_ - 1);
    int m = n >> 1;
    float ph = (float)((m * k) & (L_ - 1)) * W;
    float sn, cs; __sincosf(ph, &sn, &cs);
    dft[idx] = (n & 1) ? (__bf16)(-sn) : (__bf16)cs;
  }
  {
    int t = idx >> 7, kk = idx & 127;
    int m = kk >> 1;
    float c = (m == 0) ? (1.0f / L_) : (2.0f / L_);
    float ph = (float)((m * t) & (L_ - 1)) * W;
    float sn, cs; __sincosf(ph, &sn, &cs);
    idft[idx] = (kk & 1) ? (__bf16)(-c * sn) : (__bf16)(c * cs);
  }
}

// ===========================================================================
extern "C" void kernel_launch(void* const* d_in, const int* in_sizes, int n_in,
                              void* d_out, int out_size, void* d_ws, size_t ws_size,
                              hipStream_t stream)
{
  (void)in_sizes; (void)n_in; (void)out_size; (void)ws_size;
  const float* x_enc  = (const float*)d_in[0];
  const float* conv_w = (const float*)d_in[1];
  const float* wq     = (const float*)d_in[2];
  const float* bq     = (const float*)d_in[3];
  const float* wo     = (const float*)d_in[4];
  const float* bo     = (const float*)d_in[5];
  const float* fw_re  = (const float*)d_in[6];
  const float* fw_im  = (const float*)d_in[7];
  const float* c1     = (const float*)d_in[8];
  const float* c2     = (const float*)d_in[9];
  const float* enc_g  = (const float*)d_in[10];
  const float* enc_b  = (const float*)d_in[11];
  const float* ff_w   = (const float*)d_in[12];
  const float* ff_b   = (const float*)d_in[13];
  const float* ffn_g  = (const float*)d_in[14];
  const float* ffn_b  = (const float*)d_in[15];
  const float* reg_w  = (const float*)d_in[16];
  const float* reg_b  = (const float*)d_in[17];
  float* out = (float*)d_out;

  // ---- workspace carve (~300 MB total) ----
  uint8_t* p = (uint8_t*)d_ws;
  auto carve = [&](size_t n) -> void* { void* r = (void*)p; p += (n + 255) & ~(size_t)255; return r; };
  float*  xA    = (float*)carve((size_t)MROWS * D_ * 4);       // activations ping
  float*  xB    = (float*)carve((size_t)MROWS * D_ * 4);       // activations pong
  __bf16* xAbf  = (__bf16*)carve((size_t)MROWS * D_ * 2);      // bf16 mirror of xA
  __bf16* qT    = (__bf16*)carve((size_t)SIG * L_ * 2);        // q in [B,H*E,L]
  __bf16* XF2   = (__bf16*)carve((size_t)SIG * 128 * 2);       // spectrum [h][m][ri][b][e]
  __bf16* YF    = (__bf16*)carve((size_t)SIG * 128 * 2);       // mixed spectrum [row,128]
  __bf16* Gbf   = (__bf16*)carve((size_t)SIG * L_ * 2);        // irFFT out (== [32768,512])
  __bf16* Hbf   = (__bf16*)carve((size_t)8192 * DFF_ * 2);     // FFN hidden (row chunk)
  __bf16* wq_bf = (__bf16*)carve((size_t)2 * D_ * D_ * 2);
  __bf16* wo_bf = (__bf16*)carve((size_t)2 * D_ * D_ * 2);
  __bf16* fT_re = (__bf16*)carve((size_t)2 * H_ * MOD_ * E_ * E_ * 2);
  __bf16* fT_im = (__bf16*)carve((size_t)2 * H_ * MOD_ * E_ * E_ * 2);
  __bf16* fT_imn= (__bf16*)carve((size_t)2 * H_ * MOD_ * E_ * E_ * 2);
  __bf16* c1_bf = (__bf16*)carve((size_t)2 * DFF_ * D_ * 2);
  __bf16* c2_bf = (__bf16*)carve((size_t)2 * D_ * DFF_ * 2);
  __bf16* ffw_bf= (__bf16*)carve((size_t)D_ * D_ * 2);
  __bf16* dftb  = (__bf16*)carve((size_t)128 * L_ * 2);
  __bf16* idftb = (__bf16*)carve((size_t)L_ * 128 * 2);

  // ---- weight conversion + transposed Fourier weights + DFT bases ----
  f32_to_bf16<<<1024, 256, 0, stream>>>(wq,   wq_bf,  2 * D_ * D_);
  f32_to_bf16<<<1024, 256, 0, stream>>>(wo,   wo_bf,  2 * D_ * D_);
  f32_to_bf16<<<2048, 256, 0, stream>>>(c1,   c1_bf,  2 * DFF_ * D_);
  f32_to_bf16<<<2048, 256, 0, stream>>>(c2,   c2_bf,  2 * D_ * DFF_);
  f32_to_bf16<<<512,  256, 0, stream>>>(ff_w, ffw_bf, D_ * D_);
  fw_transpose<<<2048, 256, 0, stream>>>(fw_re, fw_im, fT_re, fT_im, fT_imn);
  build_basis<<<512, 256, 0, stream>>>(dftb, idftb);

  // ---- embedding ----
  embed<<<dim3(L_, B_), 256, 0, stream>>>(x_enc, conv_w, xA, xAbf);

  // ---- encoder layers ----
  for (int l = 0; l < 2; ++l) {
    const __bf16* wql  = wq_bf + (size_t)l * D_ * D_;
    const __bf16* wol  = wo_bf + (size_t)l * D_ * D_;
    const __bf16* frel = fT_re + (size_t)l * H_ * MOD_ * E_ * E_;
    const __bf16* fiml = fT_im + (size_t)l * H_ * MOD_ * E_ * E_;
    const __bf16* fimn = fT_imn+ (size_t)l * H_ * MOD_ * E_ * E_;
    const __bf16* c1l  = c1_bf + (size_t)l * DFF_ * D_;
    const __bf16* c2l  = c2_bf + (size_t)l * D_ * DFF_;
    const float*  bql  = bq + l * D_;
    const float*  bol  = bo + l * D_;

    // q = x @ wq^T + bq, scattered to [B,H*E,L]
    gemm_big<EP_QSCAT><<<dim3(2, 256), 256, 0, stream>>>(
        xAbf, wql, bql, nullptr, nullptr, qT, MROWS, D_, D_);
    // rFFT(64 modes) as GEMM: [16384,1024] x [1024,128] -> WMMA-native layout
    gemm_k128<EP_XSCAT><<<dim3(1, 128), 256, 0, stream>>>(
        qT, dftb, nullptr, nullptr, nullptr, XF2, SIG, 128, L_);
    // complex per-(h,m) channel mixing
    fourier_mix<<<dim3(MOD_, H_), 32, 0, stream>>>(XF2, frel, fiml, fimn, YF);
    // irFFT as GEMM: [16384,128] x [128,1024]; output == raw reshape [32768,512]
    gemm_big<EP_BF><<<dim3(4, 128), 256, 0, stream>>>(
        YF, idftb, nullptr, nullptr, nullptr, Gbf, SIG, L_, 128);
    // x = x + (fourier_out @ wo^T + bo)
    gemm_big<EP_RESID><<<dim3(2, 256), 256, 0, stream>>>(
        Gbf, wol, bol, xA, xB, nullptr, MROWS, D_, D_);
    // decomp 1
    movavg_sub<<<B_ * D_, 256, 0, stream>>>(xB, xA, xAbf);
    // FFN (chunked rows to bound workspace)
    for (int c = 0; c < 4; ++c) {
      size_t ro = (size_t)c * 8192;
      gemm_big<EP_GELU_BF><<<dim3(8, 64), 256, 0, stream>>>(
          xAbf + ro * D_, c1l, nullptr, nullptr, nullptr, Hbf, 8192, DFF_, D_);
      gemm_big<EP_RESID><<<dim3(2, 64), 256, 0, stream>>>(
          Hbf, c2l, nullptr, xA + ro * D_, xB + ro * D_, nullptr, 8192, D_, DFF_);
    }
    // decomp 2
    movavg_sub<<<B_ * D_, 256, 0, stream>>>(xB, xA, xAbf);
  }

  // ---- my_Layernorm: LN then subtract per-sequence mean ----
  ln_rows<<<MROWS, 256, 0, stream>>>(xA, enc_g, enc_b, xB);
  colmean_sub<<<dim3(2, B_), 256, 0, stream>>>(xB, xB, Gbf);
  // ---- head: Linear + LN + flatten + regression ----
  gemm_big<EP_F32><<<dim3(2, 256), 256, 0, stream>>>(
      Gbf, ffw_bf, ff_b, nullptr, xA, nullptr, MROWS, D_, D_);
  ln_rows<<<MROWS, 256, 0, stream>>>(xA, ffn_g, ffn_b, xB);
  regress<<<B_, 256, 0, stream>>>(xB, reg_w, reg_b, out);
}
